// GlobalAttention_42614665511007
// MI455X (gfx1250) — matrix operand order repaired and checked by reading, not compiled, not run on hardware
//
#include <hip/hip_runtime.h>
#include <hip/hip_bf16.h>
#include <math.h>

// ---------------- types / helpers ----------------
typedef __attribute__((ext_vector_type(16))) __bf16 bfx16;
typedef __attribute__((ext_vector_type(8)))  float  v8f;

union BFV2 { bfx16 v; uint4 q[2]; };
union U4H  { uint4 q; unsigned short h[8]; };
union U2H  { uint2 q; unsigned short h[4]; };

#define NPIX   4096
#define SEQ    256
#define SCALE  0.125f   // (512/8)^-0.5

// ---- CDNA5 async global->LDS copy (ASYNCcnt / GLOBAL_LOAD_ASYNC_TO_LDS_*) ----
#if __has_builtin(__builtin_amdgcn_global_load_async_to_lds_b128) && \
    __has_builtin(__builtin_amdgcn_global_load_async_to_lds_b64) &&  \
    __has_builtin(__builtin_amdgcn_s_wait_asynccnt)
#define USE_ASYNC 1
typedef int v4i_ __attribute__((vector_size(16)));
typedef int v2i_ __attribute__((vector_size(8)));
typedef __attribute__((address_space(1))) v4i_* g4p;
typedef __attribute__((address_space(3))) v4i_* l4p;
typedef __attribute__((address_space(1))) v2i_* g2p;
typedef __attribute__((address_space(3))) v2i_* l2p;

__device__ __forceinline__ void async_b128(const void* g, const void* l) {
  __builtin_amdgcn_global_load_async_to_lds_b128(
      (g4p)(unsigned long long)g, (l4p)(unsigned)(unsigned long long)l, 0, 0);
}
__device__ __forceinline__ void async_b64(const void* g, const void* l) {
  __builtin_amdgcn_global_load_async_to_lds_b64(
      (g2p)(unsigned long long)g, (l2p)(unsigned)(unsigned long long)l, 0, 0);
}
#define ASYNC_WAIT() __builtin_amdgcn_s_wait_asynccnt(0)
#else
#define USE_ASYNC 0
#define ASYNC_WAIT()
#endif

__device__ __forceinline__ unsigned short f2bf(float x) {
  union { float f; unsigned u; } c; c.f = x;
  unsigned r = c.u + 0x7fffu + ((c.u >> 16) & 1u);   // round-nearest-even
  return (unsigned short)(r >> 16);
}
__device__ __forceinline__ float bf2f(unsigned short b) {
  union { unsigned u; float f; } c; c.u = ((unsigned)b) << 16;
  return c.f;
}

__device__ __forceinline__ v8f wmma_bf16(bfx16 a, bfx16 b, v8f c) {
  return __builtin_amdgcn_wmma_f32_16x16x32_bf16(false, a, false, b, (short)0, c,
                                                 false, false);
}

// A-matrix 16x32 bf16 fragment from row-major [M][32] LDS tile (2x ds_load_b128).
__device__ __forceinline__ bfx16 lds_frag_a(const unsigned short* buf, int rowBase,
                                            int lane) {
  const unsigned short* rp = buf + (rowBase + (lane & 15)) * 32 + (lane >> 4) * 16;
  BFV2 r;
  r.q[0] = *(const uint4*)(rp);
  r.q[1] = *(const uint4*)(rp + 8);
  return r.v;
}
// B-matrix 32x16 fragment from TRANSPOSED [N][K] LDS tile (ldk halfs per row).
__device__ __forceinline__ bfx16 lds_frag_b_nk(const unsigned short* bT, int nBase,
                                               int ldk, int kBase, int lane) {
  const unsigned short* rp =
      bT + (nBase + (lane & 15)) * ldk + kBase + (lane >> 4) * 16;
  BFV2 r;
  r.q[0] = *(const uint4*)(rp);
  r.q[1] = *(const uint4*)(rp + 8);
  return r.v;
}
// B fragment for K^T from K stored row-major [SEQ][32].
__device__ __forceinline__ bfx16 lds_frag_kT(const unsigned short* kbuf, int keyBase,
                                             int lane) {
  const unsigned short* rp = kbuf + (keyBase + (lane & 15)) * 32 + (lane >> 4) * 16;
  BFV2 r;
  r.q[0] = *(const uint4*)(rp);
  r.q[1] = *(const uint4*)(rp + 8);
  return r.v;
}

__device__ __forceinline__ float rmax16(float x) {
  #pragma unroll
  for (int m = 1; m <= 8; m <<= 1) x = fmaxf(x, __shfl_xor(x, m, 32));
  return x;
}
__device__ __forceinline__ float rsum16(float x) {
  #pragma unroll
  for (int m = 1; m <= 8; m <<= 1) x += __shfl_xor(x, m, 32);
  return x;
}

// ---------------- prep kernels ----------------
__global__ void cvt_bf16_kernel(const float* __restrict__ src,
                                unsigned short* __restrict__ dst, int n) {
  int i = blockIdx.x * 256 + threadIdx.x;
  if (i < n) dst[i] = f2bf(src[i]);
}

// fused vertical LePE weight: Wl[o][c] = sum_m w_vert[o][m] * w_v[256+m][c]
__global__ void lepe_fuse_kernel(const float* __restrict__ wvert,
                                 const float* __restrict__ wv,
                                 unsigned short* __restrict__ dst) {
  int i = blockIdx.x * 256 + threadIdx.x;   // 256*512 threads
  int o = i >> 9, ci = i & 511;
  float acc = 0.f;
  for (int m = 0; m < 256; ++m)
    acc = fmaf(wvert[o * 256 + m], wv[(256 + m) * 512 + ci], acc);
  dst[o * 512 + ci] = f2bf(acc);
}

// ---------------- QKV (+fused LePE) projection GEMM ----------------
// Out[1792 x 4096] per batch = Wbig[1792 x 512] * fmap_b[512 x 4096]
// rows 0..1535 -> Q/K/V (bf16), rows 1536..1791 -> LePE_v (f32)
__global__ __launch_bounds__(256) void qkv_gemm_kernel(
    const unsigned short* __restrict__ wbig, const float* __restrict__ fmap,
    unsigned short* __restrict__ qkv, float* __restrict__ lepe) {
  const int tid = threadIdx.x, lane = tid & 31, wv = tid >> 5;
  const int colBase = blockIdx.x * 64;
  const int rowBase = blockIdx.y * 64;
  const int b = blockIdx.z;

  __shared__ alignas(16) unsigned short aL[64 * 32];   // [M][K] row-major
  __shared__ alignas(16) unsigned short bT[64 * 32];   // [N][K] transposed

  v8f acc0 = {}, acc1 = {};
  const int t0 = wv * 2, t1 = t0 + 1;
  const int tm0 = t0 >> 2, tn0 = t0 & 3, tm1 = t1 >> 2, tn1 = t1 & 3;
  const int ar = tid >> 2, ak = (tid & 3) * 8;         // one 16B A chunk / thread

  for (int k0 = 0; k0 < 512; k0 += 32) {
#if USE_ASYNC
    async_b128(wbig + (size_t)(rowBase + ar) * 512 + k0 + ak,
               aL + ar * 32 + ak);
#else
    *(uint4*)(aL + ar * 32 + ak) =
        *(const uint4*)(wbig + (size_t)(rowBase + ar) * 512 + k0 + ak);
#endif
    #pragma unroll
    for (int i = 0; i < 2; ++i) {
      const int flat = tid + i * 256;                  // 512 float4 chunks
      const int kk = flat >> 4, n0 = (flat & 15) * 4;
      const float* gp = fmap + ((size_t)b * 512 + k0 + kk) * NPIX + colBase + n0;
      if (k0 + 32 < 512) __builtin_prefetch(gp + (size_t)32 * NPIX, 0, 1);
      const float4 f = *(const float4*)gp;
      bT[(n0 + 0) * 32 + kk] = f2bf(f.x);
      bT[(n0 + 1) * 32 + kk] = f2bf(f.y);
      bT[(n0 + 2) * 32 + kk] = f2bf(f.z);
      bT[(n0 + 3) * 32 + kk] = f2bf(f.w);
    }
    ASYNC_WAIT();
    __syncthreads();
    acc0 = wmma_bf16(lds_frag_a(aL, tm0 * 16, lane),
                     lds_frag_b_nk(bT, tn0 * 16, 32, 0, lane), acc0);
    acc1 = wmma_bf16(lds_frag_a(aL, tm1 * 16, lane),
                     lds_frag_b_nk(bT, tn1 * 16, 32, 0, lane), acc1);
    __syncthreads();
  }
  if (rowBase < 1536) {                                // block-uniform branch
    #pragma unroll
    for (int tt = 0; tt < 2; ++tt) {
      v8f a = tt ? acc1 : acc0;
      const int tm = tt ? tm1 : tm0, tn = tt ? tn1 : tn0;
      #pragma unroll
      for (int i = 0; i < 8; ++i) {
        const int rg = rowBase + tm * 16 + (lane >> 4) * 8 + i;
        const int cg = colBase + tn * 16 + (lane & 15);
        qkv[((size_t)b * 1536 + rg) * NPIX + cg] = f2bf(a[i]);
      }
    }
  } else {
    #pragma unroll
    for (int tt = 0; tt < 2; ++tt) {
      v8f a = tt ? acc1 : acc0;
      const int tm = tt ? tm1 : tm0, tn = tt ? tn1 : tn0;
      #pragma unroll
      for (int i = 0; i < 8; ++i) {
        const int rg = rowBase - 1536 + tm * 16 + (lane >> 4) * 8 + i;
        const int cg = colBase + tn * 16 + (lane & 15);
        lepe[((size_t)b * 256 + rg) * NPIX + cg] = a[i];
      }
    }
  }
}

// ---------------- windowed attention (flash-style, WMMA) ----------------
// grid: (win=16, head=8, z=batch*2+branch). branch 0 = horizontal (64x4 windows),
// branch 1 = vertical (4x64 windows). L=256, d=32, heads=8.
__global__ __launch_bounds__(256) void attn_kernel(
    const unsigned short* __restrict__ qkv, const float* __restrict__ lepe,
    float* __restrict__ attn_out) {
  const int tid = threadIdx.x, lane = tid & 31, wv = tid >> 5;
  const int win = blockIdx.x;
  const int hd = blockIdx.y;
  const int b = blockIdx.z >> 1;
  const int branch = blockIdx.z & 1;

  __shared__ alignas(16) unsigned short qb[SEQ * 32];      // [s][dd]
  __shared__ alignas(16) unsigned short kb[SEQ * 32];      // [s][dd]
  __shared__ alignas(16) unsigned short vT[32 * SEQ];      // [dd][s] transposed
  __shared__ alignas(16) unsigned short pb[8][16 * 32];    // per-wave P, [m][k]

  const int chBase = branch * 256 + hd * 32;
  const size_t qoffb = ((size_t)b * 1536 + chBase) * NPIX;
  const size_t koffb = qoffb + (size_t)512 * NPIX;
  const size_t voffb = qoffb + (size_t)1024 * NPIX;

  if (branch) {
    // vertical: p = win*256 + s (fully contiguous runs)
    #pragma unroll
    for (int i = 0; i < 4; ++i) {
      const int flat = tid + i * 256;                 // 1024 chunks of 8 halfs
      const int dd = flat >> 5, sc = (flat & 31) * 8;
      const size_t g = (size_t)dd * NPIX + win * 256 + sc;
      U4H uq, uk;
      uq.q = *(const uint4*)(qkv + qoffb + g);
      uk.q = *(const uint4*)(qkv + koffb + g);
      #pragma unroll
      for (int j = 0; j < 8; ++j) {
        qb[(sc + j) * 32 + dd] = uq.h[j];
        kb[(sc + j) * 32 + dd] = uk.h[j];
      }
#if USE_ASYNC
      async_b128(qkv + voffb + g, vT + dd * SEQ + sc);
#else
      *(uint4*)(vT + dd * SEQ + sc) = *(const uint4*)(qkv + voffb + g);
#endif
    }
  } else {
    // horizontal: pixel = row*64 + win*4 + c, s = row*4 + c (runs of 4)
    #pragma unroll
    for (int i = 0; i < 8; ++i) {
      const int flat = tid + i * 256;                 // 2048 chunks of 4 halfs
      const int dd = flat >> 6, r = flat & 63;
      const size_t g = (size_t)dd * NPIX + r * 64 + win * 4;
      const int s0 = r * 4;
      U2H uq, uk;
      uq.q = *(const uint2*)(qkv + qoffb + g);
      uk.q = *(const uint2*)(qkv + koffb + g);
      #pragma unroll
      for (int j = 0; j < 4; ++j) {
        qb[(s0 + j) * 32 + dd] = uq.h[j];
        kb[(s0 + j) * 32 + dd] = uk.h[j];
      }
#if USE_ASYNC
      async_b64(qkv + voffb + g, vT + dd * SEQ + s0);
#else
      *(uint2*)(vT + dd * SEQ + s0) = *(const uint2*)(qkv + voffb + g);
#endif
    }
  }
  ASYNC_WAIT();
  __syncthreads();

  for (int rt = wv; rt < 16; rt += 8) {     // 2 row-tiles of 16 queries per wave
    const bfx16 aq = lds_frag_a(qb, rt * 16, lane);
    v8f o0 = {}, o1 = {};
    float mrow[8], lrow[8];
    #pragma unroll
    for (int i = 0; i < 8; ++i) { mrow[i] = -3.0e38f; lrow[i] = 0.f; }

    for (int c = 0; c < 8; ++c) {           // 32 keys per chunk
      const int kbase = c * 32;
      const bfx16 bk0 = lds_frag_kT(kb, kbase, lane);
      const bfx16 bk1 = lds_frag_kT(kb, kbase + 16, lane);
      v8f z = {};
      v8f s0 = wmma_bf16(aq, bk0, z);
      v8f s1 = wmma_bf16(aq, bk1, z);
      #pragma unroll
      for (int i = 0; i < 8; ++i) {
        const float x0 = s0[i] * SCALE, x1 = s1[i] * SCALE;
        const float tmax = rmax16(fmaxf(x0, x1));
        const float mnew = fmaxf(mrow[i], tmax);
        const float alpha = __expf(mrow[i] - mnew);
        const float p0 = __expf(x0 - mnew);
        const float p1 = __expf(x1 - mnew);
        lrow[i] = lrow[i] * alpha + rsum16(p0 + p1);
        mrow[i] = mnew;
        o0[i] *= alpha; o1[i] *= alpha;
        s0[i] = p0; s1[i] = p1;
      }
      __syncthreads();
      #pragma unroll
      for (int i = 0; i < 8; ++i) {        // C-layout -> LDS (per-wave scratch)
        const int rl = (lane >> 4) * 8 + i;
        pb[wv][rl * 32 + (lane & 15)] = f2bf(s0[i]);
        pb[wv][rl * 32 + 16 + (lane & 15)] = f2bf(s1[i]);
      }
      __syncthreads();
      const bfx16 ap = lds_frag_a(pb[wv], 0, lane);
      const bfx16 bv0 = lds_frag_b_nk(vT, 0, SEQ, kbase, lane);
      const bfx16 bv1 = lds_frag_b_nk(vT, 16, SEQ, kbase, lane);
      o0 = wmma_bf16(ap, bv0, o0);
      o1 = wmma_bf16(ap, bv1, o1);
    }

    #pragma unroll
    for (int i = 0; i < 8; ++i) {
      const float inv = 1.0f / lrow[i];
      const int ml = (lane >> 4) * 8 + i;
      const int s = rt * 16 + ml;
      const int p = branch ? (win * 256 + s) : ((s >> 2) * 64 + win * 4 + (s & 3));
      #pragma unroll
      for (int t = 0; t < 2; ++t) {
        const int dd = t * 16 + (lane & 15);
        float val = (t ? o1[i] : o0[i]) * inv;
        if (branch == 0)
          val += bf2f(vT[dd * SEQ + s]);           // lepe_h = raw v (source bug)
        else
          val += lepe[((size_t)b * 256 + hd * 32 + dd) * NPIX + p];
        attn_out[((size_t)b * 512 + chBase + dd) * NPIX + p] = val;
      }
    }
  }
}

// ---------------- output projection + residual ----------------
__global__ __launch_bounds__(256) void proj_gemm_kernel(
    const unsigned short* __restrict__ wproj, const float* __restrict__ attn,
    const float* __restrict__ fmap, float* __restrict__ out) {
  const int tid = threadIdx.x, lane = tid & 31, wv = tid >> 5;
  const int colBase = blockIdx.x * 64;
  const int rowBase = blockIdx.y * 64;
  const int b = blockIdx.z;

  __shared__ alignas(16) unsigned short aL[64 * 32];   // [M][K]
  __shared__ alignas(16) unsigned short bT[64 * 32];   // [N][K]

  v8f acc0 = {}, acc1 = {};
  const int t0 = wv * 2, t1 = t0 + 1;
  const int tm0 = t0 >> 2, tn0 = t0 & 3, tm1 = t1 >> 2, tn1 = t1 & 3;
  const int ar = tid >> 2, ak = (tid & 3) * 8;

  for (int k0 = 0; k0 < 512; k0 += 32) {
#if USE_ASYNC
    async_b128(wproj + (size_t)(rowBase + ar) * 512 + k0 + ak,
               aL + ar * 32 + ak);
#else
    *(uint4*)(aL + ar * 32 + ak) =
        *(const uint4*)(wproj + (size_t)(rowBase + ar) * 512 + k0 + ak);
#endif
    #pragma unroll
    for (int i = 0; i < 2; ++i) {
      const int flat = tid + i * 256;
      const int kk = flat >> 4, n0 = (flat & 15) * 4;
      const float* gp = attn + ((size_t)b * 512 + k0 + kk) * NPIX + colBase + n0;
      if (k0 + 32 < 512) __builtin_prefetch(gp + (size_t)32 * NPIX, 0, 1);
      const float4 f = *(const float4*)gp;
      bT[(n0 + 0) * 32 + kk] = f2bf(f.x);
      bT[(n0 + 1) * 32 + kk] = f2bf(f.y);
      bT[(n0 + 2) * 32 + kk] = f2bf(f.z);
      bT[(n0 + 3) * 32 + kk] = f2bf(f.w);
    }
    ASYNC_WAIT();
    __syncthreads();
    acc0 = wmma_bf16(lds_frag_a(aL, tm0 * 16, lane),
                     lds_frag_b_nk(bT, tn0 * 16, 32, 0, lane), acc0);
    acc1 = wmma_bf16(lds_frag_a(aL, tm1 * 16, lane),
                     lds_frag_b_nk(bT, tn1 * 16, 32, 0, lane), acc1);
    __syncthreads();
  }
  #pragma unroll
  for (int tt = 0; tt < 2; ++tt) {
    v8f a = tt ? acc1 : acc0;
    const int tm = tt ? tm1 : tm0, tn = tt ? tn1 : tn0;
    #pragma unroll
    for (int i = 0; i < 8; ++i) {
      const int rg = rowBase + tm * 16 + (lane >> 4) * 8 + i;
      const int cg = colBase + tn * 16 + (lane & 15);
      const size_t o = ((size_t)b * 512 + rg) * NPIX + cg;
      out[o] = fmap[o] + a[i];
    }
  }
}

// ---------------- launcher ----------------
extern "C" void kernel_launch(void* const* d_in, const int* in_sizes, int n_in,
                              void* d_out, int out_size, void* d_ws, size_t ws_size,
                              hipStream_t stream) {
  (void)in_sizes; (void)n_in; (void)out_size; (void)ws_size;
  const float* fmap   = (const float*)d_in[0];
  const float* w_qk   = (const float*)d_in[1];
  const float* w_v    = (const float*)d_in[2];
  const float* w_vert = (const float*)d_in[3];
  // d_in[4] = w_v_horiz: its output is discarded by the reference (source bug)
  const float* w_proj = (const float*)d_in[5];
  float* out = (float*)d_out;

  char* ws = (char*)d_ws;
  size_t off = 0;
  unsigned short* wbig = (unsigned short*)(ws + off); off += (size_t)1792 * 512 * 2;
  unsigned short* wpj  = (unsigned short*)(ws + off); off += (size_t)512 * 512 * 2;
  off = (off + 255) & ~(size_t)255;
  unsigned short* qkv  = (unsigned short*)(ws + off); off += (size_t)8 * 1536 * NPIX * 2;
  float* lepe = (float*)(ws + off); off += (size_t)8 * 256 * NPIX * 4;
  float* attn = (float*)(ws + off); off += (size_t)8 * 512 * NPIX * 4;

  cvt_bf16_kernel<<<(1024 * 512 + 255) / 256, 256, 0, stream>>>(w_qk, wbig, 1024 * 512);
  cvt_bf16_kernel<<<(512 * 512 + 255) / 256, 256, 0, stream>>>(w_v, wbig + 1024 * 512,
                                                               512 * 512);
  cvt_bf16_kernel<<<(512 * 512 + 255) / 256, 256, 0, stream>>>(w_proj, wpj, 512 * 512);
  lepe_fuse_kernel<<<(256 * 512) / 256, 256, 0, stream>>>(w_vert, w_v,
                                                          wbig + 1536 * 512);

  qkv_gemm_kernel<<<dim3(64, 28, 8), 256, 0, stream>>>(wbig, fmap, qkv, lepe);
  attn_kernel<<<dim3(16, 8, 16), 256, 0, stream>>>(qkv, lepe, attn);
  proj_gemm_kernel<<<dim3(64, 8, 8), 256, 0, stream>>>(wpj, attn, fmap, out);
}
